// Custom_LSTM_2525440770787
// MI455X (gfx1250) — matrix operand order repaired and checked by reading, not compiled, not run on hardware
//
#include <hip/hip_runtime.h>

typedef __attribute__((ext_vector_type(16))) __bf16 v16bf;
typedef __attribute__((ext_vector_type(8)))  float  v8f;
typedef unsigned short u16;
typedef unsigned int   u32;
typedef __attribute__((ext_vector_type(4)))  u32    v4u;

#define T_DIM 256
#define B_DIM 128
#define D_DIM 1024
#define H_DIM 1024
#define G4    4096  // 4*H

#if __has_builtin(__builtin_amdgcn_global_load_async_to_lds_b32) && \
    __has_builtin(__builtin_amdgcn_s_wait_asynccnt)
#define USE_ASYNC 1
typedef __attribute__((address_space(1))) int gint;
typedef __attribute__((address_space(3))) int lint;
#else
#define USE_ASYNC 0
#endif

union AU { v16bf v; u32 d[8]; };
union BU { v16bf v; u32 d[8]; };

__device__ __forceinline__ u16 f2bf(float f) {
    union { float f; u32 u; } v; v.f = f;
    u32 u = v.u;
    u32 r = u + 0x7FFFu + ((u >> 16) & 1u);  // round-to-nearest-even
    return (u16)(r >> 16);
}

__device__ __forceinline__ float sigmf(float x) { return 1.0f / (1.0f + expf(-x)); }

// ---------------- conversion / init kernels ----------------

__global__ void cvt_bf16_kernel(const float* __restrict__ src, u16* __restrict__ dst, int n) {
    int i = blockIdx.x * blockDim.x + threadIdx.x;
    if (i < n) dst[i] = f2bf(src[i]);
}

// W [K][G4] fp32 -> Wt [G4][K] bf16 (transpose via padded LDS tile, both sides coalesced)
__global__ __launch_bounds__(256) void cvt_transpose_kernel(const float* __restrict__ W,
                                                            u16* __restrict__ Wt, int K) {
    __shared__ u16 t[32][33];
    const int k0 = blockIdx.x * 32, n0 = blockIdx.y * 32;
    const int tx = threadIdx.x & 31, ty = threadIdx.x >> 5;  // 32 x 8
    #pragma unroll
    for (int i = ty; i < 32; i += 8)
        t[i][tx] = f2bf(W[(size_t)(k0 + i) * G4 + n0 + tx]);
    __syncthreads();
    #pragma unroll
    for (int i = ty; i < 32; i += 8)
        Wt[(size_t)(n0 + i) * K + k0 + tx] = t[tx][i];
}

__global__ void init_state_kernel(const float* __restrict__ csh,
                                  float* __restrict__ c, float* __restrict__ h,
                                  u16* __restrict__ hb) {
    int i = blockIdx.x * blockDim.x + threadIdx.x;
    if (i >= B_DIM * H_DIM) return;
    int b = i / H_DIM, j = i % H_DIM;
    float cv = csh[b * 2 * H_DIM + j];
    float hv = csh[b * 2 * H_DIM + H_DIM + j];
    c[i] = cv;
    h[i] = hv;
    hb[i] = f2bf(hv);
}

// ---------------- shared WMMA K-loop ----------------
// Weights are pre-transposed: Bt[G4][KTOT] bf16, so WMMA B-operand order is the
// natural memory order. Each wave owns one 16-row A slab and 4 output tiles at
// columns ncol0 + g*CSTRIDE (g=0..3), sharing the A operand across 4 WMMAs.
// Staging is chunked: 8 K-steps (32 KB) per async burst, double-buffered,
// ONE barrier per chunk; the 8-K-step compute body has no syncs at all.
// USE_ASYNC=1: chunk staged with 32x global_load_async_to_lds_b32 per thread.
// USE_ASYNC=0: chunk staged with plain global->register->ds_store copies.

#define KSTEPS_PER_CHUNK 8
#define CHUNK_U16 (KSTEPS_PER_CHUNK * 4 * 512)   // 16384 u16 = 32 KB

template<int CSTRIDE, int KTOT>
__device__ __forceinline__ void wmma_kloop(const u32* __restrict__ arow32, bool zeroA,
                                           const u16* __restrict__ Bt, int ncol0,
                                           u16* __restrict__ lds, v8f* acc) {
    const int tid  = threadIdx.x;
    const int lane = tid & 31;
    const int half = lane >> 4;
    const int nl   = lane & 15;

    // staging slot: thread copies one u32 = K pair (se,se+1) of tile column sc
    const int sc = tid >> 4;         // 0..15 column within tile
    const int se = (tid & 15) * 2;   // even K index within 32-wide K step
    const int ldst = sc * 32 + se;   // u16 index within a 512-u16 tile

    const u16* gp[4];
    #pragma unroll
    for (int g = 0; g < 4; ++g)
        gp[g] = Bt + (size_t)(ncol0 + g * CSTRIDE + sc) * KTOT + se;

    constexpr int NCH = KTOT / (32 * KSTEPS_PER_CHUNK);

    // stage chunk 0 into buffer 0
#if USE_ASYNC
    #pragma unroll
    for (int s = 0; s < KSTEPS_PER_CHUNK; ++s)
        #pragma unroll
        for (int g = 0; g < 4; ++g)
            __builtin_amdgcn_global_load_async_to_lds_b32(
                (gint*)(gp[g] + s * 32),
                (lint*)(lds + s * 2048 + g * 512 + ldst), 0, 0);
    #pragma unroll
    for (int g = 0; g < 4; ++g) gp[g] += 32 * KSTEPS_PER_CHUNK;
#endif

    const u32* ap = arow32;

    #pragma unroll 1
    for (int c = 0; c < NCH; ++c) {
        u16* buf = lds + (c & 1) * CHUNK_U16;
#if USE_ASYNC
        __builtin_amdgcn_s_wait_asynccnt(0);  // this wave's staged chunk landed
        __syncthreads();                      // everyone's chunk landed
        if (c + 1 < NCH) {                    // fire next chunk into other buffer
            u16* nbuf = lds + ((c + 1) & 1) * CHUNK_U16;
            #pragma unroll
            for (int s = 0; s < KSTEPS_PER_CHUNK; ++s)
                #pragma unroll
                for (int g = 0; g < 4; ++g)
                    __builtin_amdgcn_global_load_async_to_lds_b32(
                        (gint*)(gp[g] + s * 32),
                        (lint*)(nbuf + s * 2048 + g * 512 + ldst), 0, 0);
            #pragma unroll
            for (int g = 0; g < 4; ++g) gp[g] += 32 * KSTEPS_PER_CHUNK;
        }
#else
        __syncthreads();                      // prior chunk's readers done
        #pragma unroll
        for (int s = 0; s < KSTEPS_PER_CHUNK; ++s)
            #pragma unroll
            for (int g = 0; g < 4; ++g)
                *(u32*)&buf[s * 2048 + g * 512 + ldst] = *(const u32*)(gp[g] + s * 32);
        #pragma unroll
        for (int g = 0; g < 4; ++g) gp[g] += 32 * KSTEPS_PER_CHUNK;
        __syncthreads();                      // chunk staged
#endif
        // ---- pure compute: 8 K-steps, no syncs ----
        #pragma unroll
        for (int s = 0; s < KSTEPS_PER_CHUNK; ++s) {
            AU a;  // ISA 16-bit A layout: K pairs per VGPR, 8/half split
            #pragma unroll
            for (int q = 0; q < 8; ++q) {
                int off = s * 16 + q + ((q >= 4) ? 4 : 0) + 4 * half;
                a.d[q] = zeroA ? 0u : ap[off];
            }
            #pragma unroll
            for (int g = 0; g < 4; ++g) {
                BU b;
                const u16* src = buf + s * 2048 + g * 512 + nl * 32 + 8 * half;
                *(v4u*)&b.d[0] = *(const v4u*)src;        // ds_load_b128
                *(v4u*)&b.d[4] = *(const v4u*)(src + 16); // ds_load_b128
                acc[g] = __builtin_amdgcn_wmma_f32_16x16x32_bf16(false, a.v, false, b.v,
                                                                 (short)0, acc[g], false, false);
            }
        }
        ap += 16 * KSTEPS_PER_CHUNK;
    }
}

// ---------------- big parallel GEMM: Xg = X @ Wx + b ----------------
// grid: (G4/64, T*B/128), block 256 (8 waves = 8 row tiles; 4 col tiles per wave)

__global__ __launch_bounds__(256) void gemm_x_kernel(const u16* __restrict__ Xb,
                                                     const u16* __restrict__ Wxt,
                                                     const float* __restrict__ bias,
                                                     float* __restrict__ Xg) {
    __shared__ u16 lds[2 * CHUNK_U16];  // 64 KB double-buffered
    const int tid  = threadIdx.x;
    const int wave = tid >> 5;
    const int lane = tid & 31;
    const int half = lane >> 4;
    const int nl   = lane & 15;
    const int ncol0 = blockIdx.x * 64;
    const int row0  = (blockIdx.y * 8 + wave) * 16;
    const int arow  = row0 + nl;

    v8f acc[4] = {};
    wmma_kloop<16, D_DIM>((const u32*)(Xb + (size_t)arow * D_DIM), false, Wxt, ncol0, lds, acc);

    #pragma unroll
    for (int g = 0; g < 4; ++g) {
        const int col = ncol0 + g * 16 + nl;
        const float bv = bias[col];
        #pragma unroll
        for (int v = 0; v < 8; ++v) {  // C/D layout: M = v + 8*half
            int row = row0 + v + 8 * half;
            Xg[(size_t)row * G4 + col] = acc[g][v] + bv;
        }
    }
}

// ---------------- recurrent step: gates = Xg_t + (h*(1-m)) @ Wh; fused update ----
// grid: H/16 blocks (H-column tile), block 256 (8 waves = 8 batch-row tiles)
// the wave's 4 tiles are the 4 gates (i,f,o,g) at column stride H

__global__ __launch_bounds__(256) void lstm_step_kernel(const float* __restrict__ Xg_t,
                                                        const float* __restrict__ Xg_next,
                                                        const u16* __restrict__ Wht,
                                                        const float* __restrict__ mask_t,
                                                        const u16* __restrict__ hb_in,
                                                        u16* __restrict__ hb_out,
                                                        float* __restrict__ c_state,
                                                        float* __restrict__ h_state,
                                                        float* __restrict__ hs_t) {
    __shared__ u16 lds[2 * CHUNK_U16];  // 64 KB double-buffered
    const int tid  = threadIdx.x;
    const int wave = tid >> 5;
    const int lane = tid & 31;
    const int half = lane >> 4;
    const int nl   = lane & 15;
    const int ncol0 = blockIdx.x * 16;
    const int col   = ncol0 + nl;

    // pull next step's 2MB Xg slab toward L2 while this step computes
    // (16384 threads x 128B = exact slab coverage)
    {
        const size_t lid = (size_t)blockIdx.x * 256 + tid;
        __builtin_prefetch(Xg_next + lid * 32, 0, 1);
    }

    // init accumulators from precomputed x@Wx + b
    v8f acc[4];
    #pragma unroll
    for (int g = 0; g < 4; ++g) {
        #pragma unroll
        for (int v = 0; v < 8; ++v) {
            int row = wave * 16 + v + 8 * half;
            acc[g][v] = Xg_t[(size_t)row * G4 + g * H_DIM + col];
        }
    }

    // A operand rows: masked previous hidden state (mask is exactly 0/1)
    const int arow = wave * 16 + nl;
    const bool zeroA = (mask_t[arow] != 0.0f);
    wmma_kloop<H_DIM, H_DIM>((const u32*)(hb_in + (size_t)arow * H_DIM), zeroA,
                             Wht, ncol0, lds, acc);

    // fused element-wise LSTM update; each (row,col) owned by exactly one lane
    #pragma unroll
    for (int v = 0; v < 8; ++v) {
        const int row = wave * 16 + v + 8 * half;
        const float mrow = mask_t[row];
        const size_t idx = (size_t)row * H_DIM + col;
        float cp = c_state[idx] * (1.0f - mrow);
        float ig = sigmf(acc[0][v]);
        float fg = sigmf(acc[1][v]);
        float og = sigmf(acc[2][v]);
        float gg = tanhf(acc[3][v]);
        float cn = fg * cp + ig * gg;
        float hn = og * tanhf(cn);
        c_state[idx] = cn;
        h_state[idx] = hn;
        hb_out[idx]  = f2bf(hn);
        __builtin_nontemporal_store(hn, &hs_t[idx]);  // hs never re-read on device
    }
}

// ---------------- final concat(c, h) ----------------

__global__ void final_copy_kernel(const float* __restrict__ c, const float* __restrict__ h,
                                  float* __restrict__ out_tail) {
    int i = blockIdx.x * blockDim.x + threadIdx.x;
    if (i >= B_DIM * H_DIM) return;
    int b = i / H_DIM, j = i % H_DIM;
    out_tail[(size_t)b * 2 * H_DIM + j] = c[i];
    out_tail[(size_t)b * 2 * H_DIM + H_DIM + j] = h[i];
}

// ---------------- host launch ----------------

extern "C" void kernel_launch(void* const* d_in, const int* in_sizes, int n_in,
                              void* d_out, int out_size, void* d_ws, size_t ws_size,
                              hipStream_t stream) {
    const float* x    = (const float*)d_in[0];  // [T,B,D]
    const float* mask = (const float*)d_in[1];  // [T,B,1]
    const float* csh  = (const float*)d_in[2];  // [B,2H]
    const float* Wx   = (const float*)d_in[3];  // [D,4H]
    const float* Wh   = (const float*)d_in[4];  // [H,4H]
    const float* bias = (const float*)d_in[5];  // [4H]
    float* out = (float*)d_out;                 // hs [T,B,H] ++ [B,2H]

    char* ws = (char*)d_ws;
    u16*   Xb      = (u16*)(ws);                    //  64 MiB  x in bf16 [T*B][D]
    u16*   Wxt     = (u16*)(ws + 67108864);         //   8 MiB  Wx^T bf16 [G4][D]
    u16*   Wht     = (u16*)(ws + 75497472);         //   8 MiB  Wh^T bf16 [G4][H]
    u16*   hb0     = (u16*)(ws + 83886080);         // 256 KiB  h bf16 ping
    u16*   hb1     = (u16*)(ws + 84148224);         // 256 KiB  h bf16 pong
    float* c_state = (float*)(ws + 84410368);       // 512 KiB
    float* h_state = (float*)(ws + 84934656);       // 512 KiB
    float* Xg      = (float*)(ws + 85458944);       // 512 MiB  x@Wx + b, fp32

    {
        int n = T_DIM * B_DIM * D_DIM;
        cvt_bf16_kernel<<<(n + 255) / 256, 256, 0, stream>>>(x, Xb, n);
    }
    cvt_transpose_kernel<<<dim3(D_DIM / 32, G4 / 32), 256, 0, stream>>>(Wx, Wxt, D_DIM);
    cvt_transpose_kernel<<<dim3(H_DIM / 32, G4 / 32), 256, 0, stream>>>(Wh, Wht, H_DIM);
    init_state_kernel<<<(B_DIM * H_DIM + 255) / 256, 256, 0, stream>>>(csh, c_state, h_state, hb0);

    gemm_x_kernel<<<dim3(G4 / 64, (T_DIM * B_DIM) / 128), 256, 0, stream>>>(Xb, Wxt, bias, Xg);

    for (int t = 0; t < T_DIM; ++t) {
        const u16* hin = (t & 1) ? hb1 : hb0;
        u16*      hout = (t & 1) ? hb0 : hb1;
        const float* slab      = Xg + (size_t)t * B_DIM * G4;
        const float* slab_next = (t + 1 < T_DIM) ? slab + (size_t)B_DIM * G4 : slab;
        lstm_step_kernel<<<H_DIM / 16, 256, 0, stream>>>(
            slab, slab_next, Wht, mask + (size_t)t * B_DIM,
            hin, hout, c_state, h_state,
            out + (size_t)t * B_DIM * H_DIM);
    }

    final_copy_kernel<<<(B_DIM * H_DIM + 255) / 256, 256, 0, stream>>>(
        c_state, h_state, out + (size_t)T_DIM * B_DIM * H_DIM);
}
